// LUT_3358664425829
// MI455X (gfx1250) — matrix operand order repaired and checked by reading, not compiled, not run on hardware
//
#include <hip/hip_runtime.h>
#include <hip/hip_bf16.h>

typedef __attribute__((ext_vector_type(16))) _Float16 v16h;
typedef __attribute__((ext_vector_type(8)))  float    v8f;

#define NB 4
#define HW_IMG (1024*1024)
#define LUT_DIM 33
#define LUT_N (LUT_DIM*LUT_DIM*LUT_DIM)   /* 35937 */
#define CL_PER_B (LUT_N*3)                /* 107811 */

/* ws layout in floats */
#define OFF_DM   0        /* 4*4096 = 16384 */
#define OFF_S    16384    /* 256*16 = 4096 (cols 8..15 zero-padded) */
#define OFF_W    20480    /* 12 (+4 pad)    */
#define OFF_CL   20496    /* 431244 ; aliased with horizontal-pass temp */

/* ---------------- Stage A: antialiased bilinear downsample 1024 -> 64 ---- */
__global__ void k_down_h(const float* __restrict__ mask, float* __restrict__ tmp) {
    int tid = blockIdx.x * blockDim.x + threadIdx.x;   /* NB*1024*64 = 262144 */
    int wo = tid & 63;
    int h  = (tid >> 6) & 1023;
    int b  = tid >> 16;
    const float* row = mask + (size_t)b * HW_IMG + (size_t)h * 1024;
    float acc = 0.f, wsum = 0.f;
    int j0 = wo * 16 - 8;
    #pragma unroll
    for (int d = 0; d < 32; ++d) {
        int j = j0 + d;
        float wt = 1.f - fabsf((float)d - 15.5f) * (1.f / 16.f);
        if (j >= 0 && j < 1024) { acc += wt * row[j]; wsum += wt; }
    }
    tmp[tid] = acc / wsum;
}

__global__ void k_down_v(const float* __restrict__ tmp, float* __restrict__ dm) {
    int tid = blockIdx.x * blockDim.x + threadIdx.x;   /* NB*64*64 = 16384 */
    int x  = tid & 63;
    int yo = (tid >> 6) & 63;
    int b  = tid >> 12;
    float acc = 0.f, wsum = 0.f;
    int j0 = yo * 16 - 8;
    #pragma unroll
    for (int d = 0; d < 32; ++d) {
        int j = j0 + d;
        float wt = 1.f - fabsf((float)d - 15.5f) * (1.f / 16.f);
        if (j >= 0 && j < 1024) { acc += wt * tmp[((size_t)b * 1024 + j) * 64 + x]; wsum += wt; }
    }
    dm[tid] = acc / wsum;
}

/* ---------------- Stage B: masked feature sums  S[c][16] (padded) -------- */
__global__ void k_sreduce(const float* __restrict__ fea, const float* __restrict__ dm,
                          float* __restrict__ S) {
    __shared__ float s1[256], s2[256];
    int c = blockIdx.x & 255;
    int b = blockIdx.x >> 8;
    int t = threadIdx.x;
    const float* f = fea + ((size_t)(b * 256 + c)) * 4096;
    const float* d = dm + b * 4096;
    float sfd = 0.f, sf = 0.f;
    for (int i = t; i < 4096; i += 256) {
        float fv = f[i], dv = d[i];
        sfd += fv * dv; sf += fv;
    }
    s1[t] = sfd; s2[t] = sf;
    __syncthreads();
    for (int st = 128; st > 0; st >>= 1) {
        if (t < st) { s1[t] += s1[t + st]; s2[t] += s2[t + st]; }
        __syncthreads();
    }
    if (t == 0) {
        S[c * 16 + 2 * b]     = s1[0];            /* fg-weighted sum */
        S[c * 16 + 2 * b + 1] = s2[0] - s1[0];    /* bg-weighted sum */
    }
    /* zero the padding columns 8..15 (all 4 b-blocks write identical zeros) */
    if (t >= 8 && t < 16) S[c * 16 + t] = 0.f;
}

/* ---------------- Stage C: WMMA matvec (conv) + FC -> 12 weights ---------- */
__global__ void
__launch_bounds__(512)
k_weights(const float* __restrict__ conv_w, const float* __restrict__ conv_b,
          const float* __restrict__ fc_w,   const float* __restrict__ fc_b,
          const float* __restrict__ S,      const float* __restrict__ dm,
          float* __restrict__ Wout) {
    __shared__ float P[256 * 16];
    __shared__ float red[512];
    __shared__ float sumdm_sh[NB];

    int t    = threadIdx.x;
    int lane = t & 31;
    int w    = t >> 5;          /* wave id, 0..15 */
    int hi   = (lane >= 16);

    /* per-batch mean of dm (threads 0..127 active, EXEC restored after) */
    float part = 0.f;
    if (w < NB) {
        const float* d = dm + w * 4096;
        for (int i = 0; i < 128; ++i) part += d[lane + 32 * i];
    }
    red[t] = part;
    __syncthreads();
    if (t < NB) {
        float s = 0.f;
        for (int l = 0; l < 32; ++l) s += red[t * 32 + l];
        sumdm_sh[t] = s;
    }
    __syncthreads();

    /* P[o,j] = sum_c conv_w[o,c] * S[c,j]   (one 16x16 o-tile per wave) */
    union { v16h v; _Float16 h[16]; } A, B;
    union { v8f  v; float    f[8];  } C;
    #pragma unroll
    for (int r = 0; r < 8; ++r) C.f[r] = 0.f;

    int o = (w << 4) + (lane & 15);
    int j = lane & 15;
    #pragma unroll
    for (int k = 0; k < 8; ++k) {
        #pragma unroll
        for (int e = 0; e < 16; ++e) {
            /* A 16x32 f16 layout: K = (e<8 ? e : e+8) + (lane>=16 ? 8 : 0) */
            int ca = (k << 5) + ((e < 8) ? e : e + 8) + (hi ? 8 : 0);
            A.h[e] = (_Float16)conv_w[o * 256 + ca];
            /* B 32x16 f16 layout: K = (e<8 ? e : e+8) + (lane>=16 ? 8 : 0), N = lane&15 */
            int cb = (k << 5) + ((e < 8) ? e : e + 8) + (hi ? 8 : 0);
            B.h[e] = (_Float16)S[cb * 16 + j];    /* branch-free: padded S */
        }
        C.v = __builtin_amdgcn_wmma_f32_16x16x32_f16(
                  false, A.v, false, B.v, (short)0, C.v, false, false);
    }
    #pragma unroll
    for (int r = 0; r < 8; ++r) {
        int ol = (w << 4) + r + (hi ? 8 : 0);
        P[ol * 16 + j] = C.f[r];
    }
    __syncthreads();

    /* final FC: weights[b,k] = fc_w[k,:256].fg + fc_w[k,256:].bg + fc_b[k] */
    if (t < NB * 3) {
        int b  = t / 3;
        int kk = t - b * 3;
        float mdm = sumdm_sh[b] * (1.f / 4096.f);
        float acc = fc_b[kk];
        for (int oo = 0; oo < 256; ++oo) {
            float fg = P[oo * 16 + 2 * b]     * (1.f / 4096.f) + conv_b[oo] * mdm;
            float bg = P[oo * 16 + 2 * b + 1] * (1.f / 4096.f) + conv_b[oo] * (1.f - mdm);
            acc += fc_w[kk * 512 + oo] * fg + fc_w[kk * 512 + 256 + oo] * bg;
        }
        Wout[t] = acc;   /* t == b*3 + kk */
    }
}

/* ---------------- Stage D: per-batch combined LUT ------------------------- */
__global__ void k_mklut(const float* __restrict__ l0, const float* __restrict__ l1,
                        const float* __restrict__ l2, const float* __restrict__ W,
                        float* __restrict__ CL) {
    int tid = blockIdx.x * blockDim.x + threadIdx.x;
    if (tid >= NB * CL_PER_B) return;
    int b = tid / CL_PER_B;
    int r = tid - b * CL_PER_B;
    int i = r / 3;
    int c = r - i * 3;
    int li = c * LUT_N + i;
    CL[tid] = W[b * 3 + 0] * l0[li] + W[b * 3 + 1] * l1[li] + W[b * 3 + 2] * l2[li];
}

/* ---------------- Stage E: per-pixel single trilerp (memory-bound) -------- */
__global__ void k_apply(const float* __restrict__ img, const float* __restrict__ CL,
                        float* __restrict__ out) {
    int tid = blockIdx.x * blockDim.x + threadIdx.x;
    int hw = tid & (HW_IMG - 1);
    int b  = tid >> 20;
    size_t base = (size_t)b * 3 * HW_IMG + hw;

    float r  = __builtin_nontemporal_load(img + base);
    float g  = __builtin_nontemporal_load(img + base + HW_IMG);
    float bl = __builtin_nontemporal_load(img + base + 2 * HW_IMG);

    const float invBin = 32.f / 1.000001f;   /* 1/BINSIZE */
    float sr = r * invBin, sg = g * invBin, sb = bl * invBin;
    int ir = (int)floorf(sr); ir = ir < 0 ? 0 : (ir > 31 ? 31 : ir);
    int ig = (int)floorf(sg); ig = ig < 0 ? 0 : (ig > 31 ? 31 : ig);
    int ib = (int)floorf(sb); ib = ib < 0 ? 0 : (ib > 31 ? 31 : ib);
    float fr = sr - (float)ir;
    float fg = sg - (float)ig;
    float fb = sb - (float)ib;

    const float* L = CL + (size_t)b * CL_PER_B;
    float o0 = 0.f, o1 = 0.f, o2 = 0.f;
    #pragma unroll
    for (int dz = 0; dz < 2; ++dz) {
        float wz = dz ? fb : 1.f - fb;
        #pragma unroll
        for (int dy = 0; dy < 2; ++dy) {
            float wy = wz * (dy ? fg : 1.f - fg);
            #pragma unroll
            for (int dx = 0; dx < 2; ++dx) {
                float wv = wy * (dx ? fr : 1.f - fr);
                int idx = (((ib + dz) * 33 + (ig + dy)) * 33 + (ir + dx)) * 3;
                o0 += wv * L[idx];
                o1 += wv * L[idx + 1];
                o2 += wv * L[idx + 2];
            }
        }
    }
    __builtin_nontemporal_store(o0, out + base);
    __builtin_nontemporal_store(o1, out + base + HW_IMG);
    __builtin_nontemporal_store(o2, out + base + 2 * HW_IMG);
}

extern "C" void kernel_launch(void* const* d_in, const int* in_sizes, int n_in,
                              void* d_out, int out_size, void* d_ws, size_t ws_size,
                              hipStream_t stream) {
    const float* image  = (const float*)d_in[0];
    const float* mask   = (const float*)d_in[1];
    const float* fea    = (const float*)d_in[2];
    const float* lut0   = (const float*)d_in[3];
    const float* lut1   = (const float*)d_in[4];
    const float* lut2   = (const float*)d_in[5];
    const float* conv_w = (const float*)d_in[6];
    const float* conv_b = (const float*)d_in[7];
    const float* fc_w   = (const float*)d_in[8];
    const float* fc_b   = (const float*)d_in[9];
    float* out = (float*)d_out;
    float* ws  = (float*)d_ws;

    float* dm  = ws + OFF_DM;
    float* S   = ws + OFF_S;
    float* W   = ws + OFF_W;
    float* tmp = ws + OFF_CL;   /* horizontal-pass temp, dead before CL is written */
    float* CL  = ws + OFF_CL;

    k_down_h <<< (NB * 1024 * 64) / 256, 256, 0, stream >>> (mask, tmp);
    k_down_v <<< (NB * 64 * 64) / 256,  256, 0, stream >>> (tmp, dm);
    k_sreduce<<< NB * 256, 256, 0, stream >>> (fea, dm, S);
    k_weights<<< 1, 512, 0, stream >>> (conv_w, conv_b, fc_w, fc_b, S, dm, W);
    k_mklut  <<< (NB * CL_PER_B + 255) / 256, 256, 0, stream >>> (lut0, lut1, lut2, W, CL);
    k_apply  <<< (NB * HW_IMG) / 256, 256, 0, stream >>> (image, CL, out);
}
